// GraphAttentionLayer_37847251812421
// MI455X (gfx1250) — matrix operand order repaired and checked by reading, not compiled, not run on hardware
//
#include <hip/hip_runtime.h>

typedef __attribute__((ext_vector_type(16))) _Float16 v16h;
typedef __attribute__((ext_vector_type(8)))  float    v8f;

#define BB     8
#define NN     2048
#define FIN    256
#define FOUT   128
#define ALPHA  0.2f
#define NEG_BIG (-9.0e15f)

union HVec {
  uint4    q[2];
  _Float16 h[16];
  v16h     v;
};
static_assert(sizeof(HVec) == 32, "HVec must be 32 bytes");

// ---------------------------------------------------------------------------
// Kernel 1: WhT[b][col][n] (f16) = (X @ W)^T, WMMA f16 16x16x32, W in LDS.
// Grid: (B*N)/16 blocks of 256 threads (8 waves = 8 N-tiles of one M-tile).
// ---------------------------------------------------------------------------
__global__ __launch_bounds__(256) void k_gemm1(const float* __restrict__ X,
                                               const float* __restrict__ W,
                                               _Float16* __restrict__ WhT) {
  __shared__ alignas(16) _Float16 wT[FOUT * FIN];  // [col][k], 64 KB
  const int tid = threadIdx.x;
  for (int idx = tid; idx < FIN * FOUT; idx += 256) {
    int k = idx >> 7, c = idx & 127;
    wT[c * FIN + k] = (_Float16)W[idx];
  }
  __syncthreads();

  const int wave = tid >> 5, lane = tid & 31;
  const int lane16 = lane & 15, khalf = lane >> 4;
  const int arow = blockIdx.x * 16 + lane16;           // A-tile row (M)
  const float*    xbase = X + (size_t)arow * FIN + khalf * 8;
  const _Float16* wbase = wT + (wave * 16 + lane16) * FIN + khalf * 8;

  v8f acc = {};
#pragma unroll
  for (int kt = 0; kt < FIN / 32; ++kt) {
    const float* xp = xbase + kt * 32;
    float4 f0 = *(const float4*)(xp);
    float4 f1 = *(const float4*)(xp + 4);
    float4 f2 = *(const float4*)(xp + 16);
    float4 f3 = *(const float4*)(xp + 20);
    HVec ua;
    ua.h[0]  = (_Float16)f0.x; ua.h[1]  = (_Float16)f0.y;
    ua.h[2]  = (_Float16)f0.z; ua.h[3]  = (_Float16)f0.w;
    ua.h[4]  = (_Float16)f1.x; ua.h[5]  = (_Float16)f1.y;
    ua.h[6]  = (_Float16)f1.z; ua.h[7]  = (_Float16)f1.w;
    ua.h[8]  = (_Float16)f2.x; ua.h[9]  = (_Float16)f2.y;
    ua.h[10] = (_Float16)f2.z; ua.h[11] = (_Float16)f2.w;
    ua.h[12] = (_Float16)f3.x; ua.h[13] = (_Float16)f3.y;
    ua.h[14] = (_Float16)f3.z; ua.h[15] = (_Float16)f3.w;
    HVec ub;
    ub.q[0] = *(const uint4*)(wbase + kt * 32);
    ub.q[1] = *(const uint4*)(wbase + kt * 32 + 16);
    acc = __builtin_amdgcn_wmma_f32_16x16x32_f16(false, ua.v, false, ub.v,
                                                 (short)0, acc, false, false);
  }
  // C layout: VGPR g, lane -> M = g + 8*(lane>>4), N = lane&15
  const int col = wave * 16 + lane16;
#pragma unroll
  for (int g = 0; g < 8; ++g) {
    int row = blockIdx.x * 16 + g + 8 * khalf;         // global row (b*N + n)
    int b = row >> 11, n = row & (NN - 1);
    WhT[(size_t)b * FOUT * NN + (size_t)col * NN + n] = (_Float16)acc[g];
  }
}

// ---------------------------------------------------------------------------
// Kernel 2: s1[r] = Wh[r,:]·a1, s2[r] = Wh[r,:]·a2 (reads f16 WhT).
// ---------------------------------------------------------------------------
__global__ __launch_bounds__(256) void k_scores(const _Float16* __restrict__ WhT,
                                                const float* __restrict__ a,
                                                float* __restrict__ s1g,
                                                float* __restrict__ s2g) {
  __shared__ float a1s[FOUT], a2s[FOUT];
  const int tid = threadIdx.x;
  if (tid < FOUT) { a1s[tid] = a[tid]; a2s[tid] = a[FOUT + tid]; }
  __syncthreads();
  const int rid = blockIdx.x * 256 + tid;
  const int b = rid >> 11, n = rid & (NN - 1);
  const _Float16* base = WhT + (size_t)b * FOUT * NN + n;
  float s1 = 0.f, s2 = 0.f;
  for (int c = 0; c < FOUT; ++c) {
    float w = (float)base[(size_t)c * NN];
    s1 += w * a1s[c];
    s2 += w * a2s[c];
  }
  s1g[rid] = s1;
  s2g[rid] = s2;
}

// ---------------------------------------------------------------------------
// Kernel 3: per 16-row block: online softmax stats over A, p->f16 in LDS,
// then out = (P @ Wh) / rowsum via WMMA. Grid: (N/16, B) x 256 threads.
// ---------------------------------------------------------------------------
__global__ __launch_bounds__(256) void k_attn(const float* __restrict__ A,
                                              const _Float16* __restrict__ WhT,
                                              const float* __restrict__ s1g,
                                              const float* __restrict__ s2g,
                                              float* __restrict__ out) {
  __shared__ alignas(16) _Float16 p_ls[16 * NN];       // 64 KB
  __shared__ float s2s[NN];                            // 8 KB
  __shared__ float red_m[256], red_s[256];
  __shared__ float rowmax[16], rowsum[16], s1s[16];

  const int tid = threadIdx.x;
  const int b  = blockIdx.y;
  const int m0 = blockIdx.x * 16;
  const size_t abase = ((size_t)b * NN + m0) * NN;
  const int rowbase = b * NN + m0;
  const int nbase   = b * NN;

  // Cache s2 row for this batch (read twice below)
  for (int n = tid; n < NN; n += 256) s2s[n] = s2g[nbase + n];
  __syncthreads();

  // Phase A: per-thread online (max, sumexp) over strided columns
  {
    const int r = tid >> 4, j0 = tid & 15;
    const float s1v = s1g[rowbase + r];
    const float* arow = A + abase + (size_t)r * NN;
    float mt = -3.0e38f, st = 0.f;
    for (int n = j0; n < NN; n += 16) {
      float av = arow[n];
      float e = s1v + s2s[n];
      e = e > 0.f ? e : ALPHA * e;
      e += NEG_BIG * (1.f - av);
      if (e <= mt) {
        st += __expf(e - mt);
      } else {
        st = st * __expf(mt - e) + 1.f;
        mt = e;
      }
    }
    red_m[tid] = mt;
    red_s[tid] = st;
  }
  __syncthreads();
  if (tid < 16) {
    float M = -3.0e38f;
    for (int i = 0; i < 16; ++i) M = fmaxf(M, red_m[tid * 16 + i]);
    float S = 0.f;
    for (int i = 0; i < 16; ++i) S += red_s[tid * 16 + i] * __expf(red_m[tid * 16 + i] - M);
    rowmax[tid] = M;
    rowsum[tid] = S;
    s1s[tid] = s1g[rowbase + tid];
  }
  __syncthreads();

  // Phase B: unnormalized p = exp(e - rowmax) as f16 in LDS (A re-read is L2-hot)
  for (int idx = tid; idx < 16 * NN; idx += 256) {
    int m = idx >> 11, n = idx & (NN - 1);
    float av = A[abase + idx];
    float e = s1s[m] + s2s[n];
    e = e > 0.f ? e : ALPHA * e;
    e += NEG_BIG * (1.f - av);
    p_ls[idx] = (_Float16)__expf(e - rowmax[m]);
  }
  __syncthreads();

  // Phase C: (P @ Wh) with WMMA; A-tile from LDS, B-tile = WhT from global
  const int wave = tid >> 5, lane = tid & 31;
  const int lane16 = lane & 15, khalf = lane >> 4;
  const _Float16* pbase = p_ls + lane16 * NN + khalf * 8;
  const _Float16* wbase = WhT + (size_t)b * FOUT * NN +
                          (size_t)(wave * 16 + lane16) * NN + khalf * 8;
  v8f acc = {};
#pragma unroll 8
  for (int kt = 0; kt < NN / 32; ++kt) {
    HVec ua, ub;
    ua.q[0] = *(const uint4*)(pbase + kt * 32);        // K lo half (8 halves)
    ua.q[1] = *(const uint4*)(pbase + kt * 32 + 16);   // K hi half
    ub.q[0] = *(const uint4*)(wbase + kt * 32);        // K lo half (8 halves)
    ub.q[1] = *(const uint4*)(wbase + kt * 32 + 16);   // K hi half
    acc = __builtin_amdgcn_wmma_f32_16x16x32_f16(false, ua.v, false, ub.v,
                                                 (short)0, acc, false, false);
  }
  const int col = wave * 16 + lane16;
#pragma unroll
  for (int g = 0; g < 8; ++g) {
    int m = g + 8 * khalf;
    out[((size_t)(b * NN) + m0 + m) * FOUT + col] = acc[g] / rowsum[m];
  }
}

// ---------------------------------------------------------------------------
extern "C" void kernel_launch(void* const* d_in, const int* in_sizes, int n_in,
                              void* d_out, int out_size, void* d_ws, size_t ws_size,
                              hipStream_t stream) {
  (void)in_sizes; (void)n_in; (void)out_size; (void)ws_size;
  const float* X = (const float*)d_in[0];   // (B, N, F_IN)
  const float* A = (const float*)d_in[1];   // (B, N, N)
  const float* W = (const float*)d_in[2];   // (F_IN, F_OUT)
  const float* a = (const float*)d_in[3];   // (2*F_OUT, 1)
  float* out = (float*)d_out;               // (B, N, F_OUT)

  char* ws = (char*)d_ws;
  _Float16* WhT = (_Float16*)ws;                                  // 4 MB
  float* s1 = (float*)(ws + (size_t)BB * FOUT * NN * sizeof(_Float16));
  float* s2 = s1 + BB * NN;

  k_gemm1 <<<(BB * NN) / 16,  256, 0, stream>>>(X, W, WhT);
  k_scores<<<(BB * NN) / 256, 256, 0, stream>>>(WhT, a, s1, s2);
  k_attn  <<<dim3(NN / 16, BB), 256, 0, stream>>>(A, WhT, s1, s2, out);
}
